// Model_75033078661865
// MI455X (gfx1250) — matrix operand order repaired and checked by reading, not compiled, not run on hardware
//
#include <hip/hip_runtime.h>
#include <hip/hip_bf16.h>
#include <math.h>

typedef _Float16 half_t;
typedef __attribute__((ext_vector_type(8)))  _Float16 v8h;
typedef __attribute__((ext_vector_type(16))) _Float16 v16h;
typedef __attribute__((ext_vector_type(8)))  float    v8f;

#define HB 16            // batch
#define HF 20            // features
#define HS 512           // sequence
#define HH 512           // hidden
#define HD 256           // MSA head dim
#define HN 513           // tokens incl. class token
#define HBT (HB * HN)    // 8208 rows (multiple of 16)
#define NPAD 544         // padded N for scores (17*32)
#define NRB 9            // rest blocks

__device__ __forceinline__ float gelu_exact(float x) {
  return 0.5f * x * (1.0f + erff(x * 0.70710678118654752440f));
}

// ---------------------------------------------------------------------------
// Batched WMMA GEMM: Y[z][m,n] = act(alpha * sum_k X[z][m,k]*W[z][n,k] + bias)
//                                (+ res[z][m,n])
// z decomposes as (zb, zh) with zh = z % NH.
// One wave computes a 32x64 strip: 2 A row-fragments x 4 B col-fragments
// -> 8 v_wmma per k-step sharing 12 b128 loads, loads issued before the
//    WMMA burst so only one s_wait_loadcnt per k-step is needed.
// ---------------------------------------------------------------------------
__global__ void __launch_bounds__(256) k_gemm(
    const half_t* __restrict__ X, long sxb, long sxh, int ldx,
    const half_t* __restrict__ W, long swb, long swh, int ldw,
    const float* __restrict__ bias, long sbh,
    const float* __restrict__ res, long srb, long srh, int ldr,
    float* __restrict__ yf, half_t* __restrict__ yh, long syb, long syh, int ldy,
    int M, int Nc, int K, int NH, float alpha, int act)
{
  const int tilesN = (Nc + 63) >> 6;
  const int tilesM = (M + 31) >> 5;
  const int wid  = threadIdx.x >> 5;
  const int lane = threadIdx.x & 31;
  const int tile = blockIdx.x * 8 + wid;
  if (tile >= tilesM * tilesN) return;              // wave-uniform exit
  const int m0 = (tile / tilesN) << 5;
  const int n0 = (tile % tilesN) << 6;
  const int z  = blockIdx.z;
  const int zb = z / NH, zh = z - zb * NH;

  const half_t* Xz = X + (size_t)zb * sxb + (size_t)zh * sxh;
  const half_t* Wz = W + (size_t)zb * swb + (size_t)zh * swh;

  // A fragment: lane<16 holds row m, K in {0..15}; lane>=16 -> K+16
  int ar0 = m0 + (lane & 15);      if (ar0 >= M) ar0 = M - 1;  // clamp: EXEC stays full
  int ar1 = m0 + 16 + (lane & 15); if (ar1 >= M) ar1 = M - 1;
  const int kA = (lane & 16) ? 16 : 0;
  const int kB = (lane & 16) ? 16 : 0;
  const half_t* xp0 = Xz + (size_t)ar0 * ldx + kA;
  const half_t* xp1 = Xz + (size_t)ar1 * ldx + kA;
  const half_t* wp[4];
#pragma unroll
  for (int j = 0; j < 4; ++j) {
    int bc = n0 + j * 16 + (lane & 15); if (bc >= Nc) bc = Nc - 1;
    wp[j] = Wz + (size_t)bc * ldw + kB;   // B: lane holds 16 contiguous K of one W row
  }

  v8f acc[8] = {};
  for (int k = 0; k < K; k += 32) {
    // --- issue all loads for this k-step first (one clause, one wait) ---
    v8h a00 = *(const v8h*)(xp0 + k);
    v8h a01 = *(const v8h*)(xp0 + k + 8);
    v8h a10 = *(const v8h*)(xp1 + k);
    v8h a11 = *(const v8h*)(xp1 + k + 8);
    v16h b0 = *(const v16h*)(wp[0] + k);   // 32B contiguous, 32B aligned by construction
    v16h b1 = *(const v16h*)(wp[1] + k);
    v16h b2 = *(const v16h*)(wp[2] + k);
    v16h b3 = *(const v16h*)(wp[3] + k);
    v16h A0 = __builtin_shufflevector(a00, a01, 0,1,2,3,4,5,6,7,8,9,10,11,12,13,14,15);
    v16h A1 = __builtin_shufflevector(a10, a11, 0,1,2,3,4,5,6,7,8,9,10,11,12,13,14,15);
    // --- WMMA burst ---
    acc[0] = __builtin_amdgcn_wmma_f32_16x16x32_f16(false, A0, false, b0, (short)0, acc[0], false, false);
    acc[1] = __builtin_amdgcn_wmma_f32_16x16x32_f16(false, A0, false, b1, (short)0, acc[1], false, false);
    acc[2] = __builtin_amdgcn_wmma_f32_16x16x32_f16(false, A0, false, b2, (short)0, acc[2], false, false);
    acc[3] = __builtin_amdgcn_wmma_f32_16x16x32_f16(false, A0, false, b3, (short)0, acc[3], false, false);
    acc[4] = __builtin_amdgcn_wmma_f32_16x16x32_f16(false, A1, false, b0, (short)0, acc[4], false, false);
    acc[5] = __builtin_amdgcn_wmma_f32_16x16x32_f16(false, A1, false, b1, (short)0, acc[5], false, false);
    acc[6] = __builtin_amdgcn_wmma_f32_16x16x32_f16(false, A1, false, b2, (short)0, acc[6], false, false);
    acc[7] = __builtin_amdgcn_wmma_f32_16x16x32_f16(false, A1, false, b3, (short)0, acc[7], false, false);
  }

  // C/D layout: VGPR r -> row (lanes 0-15) / row+8 (lanes 16-31), col n0+(lane&15)
  const int rbase = m0 + ((lane & 16) ? 8 : 0);
  float*  yfz = yf  ? yf  + (size_t)zb * syb + (size_t)zh * syh : (float*)0;
  half_t* yhz = yh  ? yh  + (size_t)zb * syb + (size_t)zh * syh : (half_t*)0;
  const float* rz = res ? res + (size_t)zb * srb + (size_t)zh * srh : (const float*)0;

#pragma unroll
  for (int j = 0; j < 4; ++j) {
    const int ccol = n0 + j * 16 + (lane & 15);
    if (ccol >= Nc) continue;
    const float bv = bias ? bias[(size_t)zh * sbh + ccol] : 0.f;
#pragma unroll
    for (int f = 0; f < 2; ++f) {              // the two 16-row A fragments
#pragma unroll
      for (int r = 0; r < 8; ++r) {
        const int crow = rbase + f * 16 + r;
        if (crow >= M) continue;
        float v = acc[f * 4 + j][r] * alpha + bv;
        if (act == 1) v = gelu_exact(v);
        if (rz)  v += rz[(size_t)crow * ldr + ccol];
        if (yfz) yfz[(size_t)crow * ldy + ccol] = v;
        if (yhz) yhz[(size_t)crow * ldy + ccol] = (half_t)v;
      }
    }
  }
}

// ---------------------------------------------------------------------------
// LayerNorm: fp32 in [rows,512] -> f16 out. One wave per row (16 elems/lane).
// ---------------------------------------------------------------------------
__global__ void __launch_bounds__(256) k_layernorm(
    const float* __restrict__ x, const float* __restrict__ g,
    const float* __restrict__ b, half_t* __restrict__ out, int rows)
{
  const int wid = threadIdx.x >> 5, lane = threadIdx.x & 31;
  const int row = blockIdx.x * 8 + wid;
  if (row >= rows) return;
  const float* xr = x + (size_t)row * HH;
  float v[16]; float s = 0.f;
#pragma unroll
  for (int i = 0; i < 16; ++i) { v[i] = xr[lane + i * 32]; s += v[i]; }
#pragma unroll
  for (int o = 16; o > 0; o >>= 1) s += __shfl_xor(s, o, 32);
  const float mean = s * (1.f / 512.f);
  float q = 0.f;
#pragma unroll
  for (int i = 0; i < 16; ++i) { float d = v[i] - mean; q += d * d; }
#pragma unroll
  for (int o = 16; o > 0; o >>= 1) q += __shfl_xor(q, o, 32);
  const float rstd = rsqrtf(q * (1.f / 512.f) + 1e-5f);
  half_t* orow = out + (size_t)row * HH;
#pragma unroll
  for (int i = 0; i < 16; ++i) {
    const int idx = lane + i * 32;
    orow[idx] = (half_t)((v[i] - mean) * rstd * g[idx] + b[idx]);
  }
}

// ---------------------------------------------------------------------------
// Softmax over rows of [Z][NPAD,NPAD] f16 scores (valid cols 0..HN-1), in place.
// Pad columns HN..NPAD-1 are written as 0 so the P@V GEMM can run K=NPAD.
// ---------------------------------------------------------------------------
__global__ void __launch_bounds__(256) k_softmax(half_t* __restrict__ sc, int Z)
{
  const int wid = threadIdx.x >> 5, lane = threadIdx.x & 31;
  const int rid = blockIdx.x * 8 + wid;
  if (rid >= Z * HN) return;
  const int z = rid / HN, m = rid - z * HN;
  half_t* row = sc + (size_t)z * NPAD * NPAD + (size_t)m * NPAD;
  float v[17]; float mx = -1e30f;
#pragma unroll
  for (int i = 0; i < 17; ++i) {
    const int idx = lane + i * 32;
    const float t = (idx < HN) ? (float)row[idx] : -1e30f;
    v[i] = t; mx = fmaxf(mx, t);
  }
#pragma unroll
  for (int o = 16; o > 0; o >>= 1) mx = fmaxf(mx, __shfl_xor(mx, o, 32));
  float s = 0.f;
#pragma unroll
  for (int i = 0; i < 17; ++i) { v[i] = expf(v[i] - mx); s += v[i]; }
#pragma unroll
  for (int o = 16; o > 0; o >>= 1) s += __shfl_xor(s, o, 32);
  const float inv = 1.f / s;
#pragma unroll
  for (int i = 0; i < 17; ++i) {
    const int idx = lane + i * 32;                   // 17*32 = 544 covers NPAD exactly
    row[idx] = (half_t)((idx < HN) ? v[i] * inv : 0.f);
  }
}

// Vt[z][d, np] = (np < HN) ? V[z][np, d] : 0   (so P@V uses the X@W^T GEMM form)
__global__ void __launch_bounds__(256) k_transpose(
    const half_t* __restrict__ v, long svb, long svh, int ldv,
    half_t* __restrict__ vt, long zstride, int D, int NH, int total)
{
  const int t = blockIdx.x * 256 + threadIdx.x;
  if (t >= total) return;
  const int np = t % NPAD; int rest = t / NPAD;
  const int d = rest % D;  const int z = rest / D;
  const int zb = z / NH, zh = z - zb * NH;
  half_t val = (half_t)0.f;
  if (np < HN) val = v[(size_t)zb * svb + (size_t)zh * svh + (size_t)np * ldv + d];
  vt[(size_t)z * zstride + (size_t)d * NPAD + np] = val;
}

// Embed: out[b, n, h] = n==0 ? cls[h] : (mapper_w[h,:].x[b,:,n-1] + mb[h]) * pe[n-1,h]
__global__ void __launch_bounds__(256) k_embed(
    const float* __restrict__ x, const float* __restrict__ mw,
    const float* __restrict__ mb, const float* __restrict__ cls,
    float* __restrict__ out)
{
  const int t = blockIdx.x * 256 + threadIdx.x;
  if (t >= HB * HN * HH) return;
  const int h = t % HH; int rest = t / HH;
  const int n = rest % HN; const int b = rest / HN;
  float val;
  if (n == 0) {
    val = cls[h];
  } else {
    const int s = n - 1;
    float acc = mb[h];
#pragma unroll
    for (int f = 0; f < HF; ++f)
      acc += x[((size_t)b * HF + f) * HS + s] * mw[h * HF + f];
    const float expo = (float)(h - (h & 1)) / (float)HH;
    const float ang = (float)s / powf(10000.f, expo);
    const float pe = (h & 1) ? cosf(ang) : sinf(ang);
    val = acc * pe;
  }
  out[t] = val;
}

__global__ void __launch_bounds__(256) k_f2h(
    const float* __restrict__ in, half_t* __restrict__ out, int n)
{
  const int t = blockIdx.x * 256 + threadIdx.x;
  if (t < n) out[t] = (half_t)in[t];
}

// Head: prob = sigmoid(out[b*HN, :].head_w + head_b); y[0:16]=rint(prob), y[16:32]=prob
__global__ void __launch_bounds__(512) k_head(
    const float* __restrict__ out, const float* __restrict__ hw,
    const float* __restrict__ hb, float* __restrict__ y)
{
  const int wid = threadIdx.x >> 5, lane = threadIdx.x & 31;   // 16 waves = 16 batches
  const float* row = out + (size_t)wid * HN * HH;
  float s = 0.f;
#pragma unroll
  for (int i = 0; i < 16; ++i) { const int idx = lane + i * 32; s += row[idx] * hw[idx]; }
#pragma unroll
  for (int o = 16; o > 0; o >>= 1) s += __shfl_xor(s, o, 32);
  if (lane == 0) {
    const float p = 1.f / (1.f + expf(-(s + hb[0])));
    y[wid] = rintf(p);          // round-half-even, matches jnp.round
    y[16 + wid] = p;
  }
}

// ---------------------------------------------------------------------------
static void launch_gemm(hipStream_t stream,
    const half_t* X, long sxb, long sxh, int ldx,
    const half_t* W, long swb, long swh, int ldw,
    const float* bias, long sbh,
    const float* res, long srb, long srh, int ldr,
    float* yf, half_t* yh, long syb, long syh, int ldy,
    int M, int Nc, int K, int Z, int NH, float alpha, int act)
{
  const int tiles = ((M + 31) / 32) * ((Nc + 63) / 64);
  dim3 grid((tiles + 7) / 8, 1, Z);
  k_gemm<<<grid, dim3(256), 0, stream>>>(X, sxb, sxh, ldx, W, swb, swh, ldw,
      bias, sbh, res, srb, srh, ldr, yf, yh, syb, syh, ldy, M, Nc, K, NH, alpha, act);
}

extern "C" void kernel_launch(void* const* d_in, const int* in_sizes, int n_in,
                              void* d_out, int out_size, void* d_ws, size_t ws_size,
                              hipStream_t stream)
{
  (void)in_sizes; (void)n_in; (void)out_size; (void)ws_size;
  const float* x_org    = (const float*)d_in[0];
  const float* x_aug    = (const float*)d_in[1];
  const float* mapper_w = (const float*)d_in[2];
  const float* mapper_b = (const float*)d_in[3];
  const float* cls_tok  = (const float*)d_in[4];
  const float* b0_ln1_g = (const float*)d_in[5];
  const float* b0_ln1_b = (const float*)d_in[6];
  const float* b0_wq    = (const float*)d_in[7];
  const float* b0_bq    = (const float*)d_in[8];
  const float* b0_wk    = (const float*)d_in[9];
  const float* b0_bk    = (const float*)d_in[10];
  const float* b0_wv    = (const float*)d_in[11];
  const float* b0_bv    = (const float*)d_in[12];
  const float* b0_pw    = (const float*)d_in[13];
  const float* b0_pb    = (const float*)d_in[14];
  const float* b0_ln2_g = (const float*)d_in[15];
  const float* b0_ln2_b = (const float*)d_in[16];
  const float* b0_w1    = (const float*)d_in[17];
  const float* b0_b1    = (const float*)d_in[18];
  const float* b0_w2    = (const float*)d_in[19];
  const float* b0_b2    = (const float*)d_in[20];
  const float* r_ln1_g  = (const float*)d_in[21];
  const float* r_ln1_b  = (const float*)d_in[22];
  const float* r_wq     = (const float*)d_in[23];
  const float* r_bq     = (const float*)d_in[24];
  const float* r_wk     = (const float*)d_in[25];
  const float* r_bk     = (const float*)d_in[26];
  const float* r_wv     = (const float*)d_in[27];
  const float* r_bv     = (const float*)d_in[28];
  const float* r_ln2_g  = (const float*)d_in[29];
  const float* r_ln2_b  = (const float*)d_in[30];
  const float* r_w1     = (const float*)d_in[31];
  const float* r_b1     = (const float*)d_in[32];
  const float* r_w2     = (const float*)d_in[33];
  const float* r_b2     = (const float*)d_in[34];
  const float* head_w   = (const float*)d_in[35];
  const float* head_b   = (const float*)d_in[36];

  char* ws = (char*)d_ws;
  size_t off = 0;
  auto alloc = [&](size_t bytes) -> void* {
    void* p = (void*)(ws + off);
    off = (off + bytes + 255) & ~(size_t)255;
    return p;
  };
  float*  outf  = (float*) alloc((size_t)HBT * HH * 4);         // residual stream
  float*  augf  = (float*) alloc((size_t)HBT * HH * 4);         // aug stream (block0 only)
  half_t* ln16  = (half_t*)alloc((size_t)HBT * HH * 2);
  half_t* lnA16 = (half_t*)alloc((size_t)HBT * HH * 2);
  half_t* q16   = (half_t*)alloc((size_t)HBT * HH * 2);
  half_t* k16   = (half_t*)alloc((size_t)HBT * HH * 2);
  half_t* v16   = (half_t*)alloc((size_t)HBT * HH * 2);
  half_t* sc16  = (half_t*)alloc((size_t)32 * NPAD * NPAD * 2); // scores/probs
  half_t* vt16  = (half_t*)alloc((size_t)16 * HH * NPAD * 2);   // V transposed
  half_t* att16 = (half_t*)alloc((size_t)HBT * 2 * HH * 2);     // block0 concat
  half_t* mh16  = (half_t*)alloc((size_t)HBT * 4 * HH * 2);     // MLP hidden
  half_t* wbuf  = (half_t*)alloc((size_t)4194304 * 2);          // f16 weight scratch

  auto cvt = [&](const float* src, half_t* dst, int n) {
    k_f2h<<<(n + 255) / 256, 256, 0, stream>>>(src, dst, n);
  };
  auto lnorm = [&](const float* x, const float* g, const float* b, half_t* o, int rows) {
    k_layernorm<<<(rows + 7) / 8, 256, 0, stream>>>(x, g, b, o, rows);
  };

  // ---- embed ----
  const int ET = HB * HN * HH;
  k_embed<<<(ET + 255) / 256, 256, 0, stream>>>(x_org, mapper_w, mapper_b, cls_tok, outf);
  k_embed<<<(ET + 255) / 256, 256, 0, stream>>>(x_aug, mapper_w, mapper_b, cls_tok, augf);

  // ---- block 0: AugHeadAttention ----
  half_t* wq16 = wbuf;
  half_t* wk16 = wbuf + 1 * 524288;
  half_t* wv16 = wbuf + 2 * 524288;
  half_t* pw16 = wbuf + 3 * 524288;
  half_t* w116 = wbuf + 4 * 524288;
  half_t* w216 = wbuf + 6 * 524288;
  cvt(b0_wq, wq16, 2 * HH * HH);
  cvt(b0_wk, wk16, 2 * HH * HH);
  cvt(b0_wv, wv16, 2 * HH * HH);
  cvt(b0_pw, pw16, 2 * HH * HH);
  cvt(b0_w1, w116, 4 * HH * HH);
  cvt(b0_w2, w216, 4 * HH * HH);
  lnorm(outf, b0_ln1_g, b0_ln1_b, ln16, HBT);
  lnorm(augf, b0_ln1_g, b0_ln1_b, lnA16, HBT);

  const float scale0 = 0.04419417382415922f;   // 1/sqrt(512)
  for (int h = 0; h < 2; ++h) {
    const half_t* xln = h ? lnA16 : ln16;
    launch_gemm(stream, xln, 0, 0, HH, wq16 + (size_t)h * HH * HH, 0, 0, HH,
                b0_bq + h * HH, 0, nullptr, 0, 0, 0,
                nullptr, q16, 0, 0, HH, HBT, HH, HH, 1, 1, 1.f, 0);
    launch_gemm(stream, xln, 0, 0, HH, wk16 + (size_t)h * HH * HH, 0, 0, HH,
                b0_bk + h * HH, 0, nullptr, 0, 0, 0,
                nullptr, k16, 0, 0, HH, HBT, HH, HH, 1, 1, 1.f, 0);
    launch_gemm(stream, xln, 0, 0, HH, wv16 + (size_t)h * HH * HH, 0, 0, HH,
                b0_bv + h * HH, 0, nullptr, 0, 0, 0,
                nullptr, v16, 0, 0, HH, HBT, HH, HH, 1, 1, 1.f, 0);
    // scores = scale * Q K^T, per batch
    launch_gemm(stream, q16, (long)HN * HH, 0, HH, k16, (long)HN * HH, 0, HH,
                nullptr, 0, nullptr, 0, 0, 0,
                nullptr, sc16, (long)NPAD * NPAD, 0, NPAD,
                HN, HN, HH, HB, 1, scale0, 0);
    k_softmax<<<((HB * HN) + 7) / 8, 256, 0, stream>>>(sc16, HB);
    { const int total = HB * HH * NPAD;
      k_transpose<<<(total + 255) / 256, 256, 0, stream>>>(
          v16, (long)HN * HH, 0, HH, vt16, (long)HH * NPAD, HH, 1, total); }
    // att = P V  -> concat columns [h*H, h*H+512)
    launch_gemm(stream, sc16, (long)NPAD * NPAD, 0, NPAD,
                vt16, (long)HH * NPAD, 0, NPAD,
                nullptr, 0, nullptr, 0, 0, 0,
                nullptr, att16 + (size_t)h * HH, (long)HN * 2 * HH, 0, 2 * HH,
                HN, HH, NPAD, HB, 1, 1.f, 0);
  }
  // out = o + proj(cat)
  launch_gemm(stream, att16, 0, 0, 2 * HH, pw16, 0, 0, 2 * HH,
              b0_pb, 0, outf, 0, 0, HH,
              outf, nullptr, 0, 0, HH, HBT, HH, 2 * HH, 1, 1, 1.f, 0);
  lnorm(outf, b0_ln2_g, b0_ln2_b, ln16, HBT);
  launch_gemm(stream, ln16, 0, 0, HH, w116, 0, 0, HH, b0_b1, 0,
              nullptr, 0, 0, 0, nullptr, mh16, 0, 0, 4 * HH,
              HBT, 4 * HH, HH, 1, 1, 1.f, 1);
  launch_gemm(stream, mh16, 0, 0, 4 * HH, w216, 0, 0, 4 * HH, b0_b2, 0,
              outf, 0, 0, HH, outf, nullptr, 0, 0, HH,
              HBT, HH, 4 * HH, 1, 1, 1.f, 0);

  // ---- blocks 1..9: MultiHeadAttention (2 heads of dim 256) ----
  for (int ib = 0; ib < NRB; ++ib) {
    half_t* rq16  = wbuf;
    half_t* rk16  = wbuf + 2 * HD * HD;
    half_t* rv16  = wbuf + 4 * HD * HD;
    half_t* rw116 = wbuf + 6 * HD * HD;
    half_t* rw216 = rw116 + 4 * HH * HH;
    cvt(r_wq + (size_t)ib * 2 * HD * HD, rq16, 2 * HD * HD);
    cvt(r_wk + (size_t)ib * 2 * HD * HD, rk16, 2 * HD * HD);
    cvt(r_wv + (size_t)ib * 2 * HD * HD, rv16, 2 * HD * HD);
    cvt(r_w1 + (size_t)ib * 4 * HH * HH, rw116, 4 * HH * HH);
    cvt(r_w2 + (size_t)ib * 4 * HH * HH, rw216, 4 * HH * HH);
    lnorm(outf, r_ln1_g + ib * HH, r_ln1_b + ib * HH, ln16, HBT);
    // q/k/v: z=(b,h), X slice = ln[b*N.., h*D..], Y = [B,2,N,D]
    launch_gemm(stream, ln16, (long)HN * HH, HD, HH, rq16, 0, (long)HD * HD, HD,
                r_bq + (size_t)ib * 2 * HD, HD, nullptr, 0, 0, 0,
                nullptr, q16, (long)2 * HN * HD, (long)HN * HD, HD,
                HN, HD, HD, 2 * HB, 2, 1.f, 0);
    launch_gemm(stream, ln16, (long)HN * HH, HD, HH, rk16, 0, (long)HD * HD, HD,
                r_bk + (size_t)ib * 2 * HD, HD, nullptr, 0, 0, 0,
                nullptr, k16, (long)2 * HN * HD, (long)HN * HD, HD,
                HN, HD, HD, 2 * HB, 2, 1.f, 0);
    launch_gemm(stream, ln16, (long)HN * HH, HD, HH, rv16, 0, (long)HD * HD, HD,
                r_bv + (size_t)ib * 2 * HD, HD, nullptr, 0, 0, 0,
                nullptr, v16, (long)2 * HN * HD, (long)HN * HD, HD,
                HN, HD, HD, 2 * HB, 2, 1.f, 0);
    launch_gemm(stream, q16, (long)2 * HN * HD, (long)HN * HD, HD,
                k16, (long)2 * HN * HD, (long)HN * HD, HD,
                nullptr, 0, nullptr, 0, 0, 0,
                nullptr, sc16, (long)2 * NPAD * NPAD, (long)NPAD * NPAD, NPAD,
                HN, HN, HD, 2 * HB, 2, 0.0625f, 0);
    k_softmax<<<((2 * HB * HN) + 7) / 8, 256, 0, stream>>>(sc16, 2 * HB);
    { const int total = 2 * HB * HD * NPAD;
      k_transpose<<<(total + 255) / 256, 256, 0, stream>>>(
          v16, (long)2 * HN * HD, (long)HN * HD, HD, vt16, (long)HD * NPAD, HD, 2, total); }
    // out += P V  (written straight into the fp32 residual stream at col h*D)
    launch_gemm(stream, sc16, (long)2 * NPAD * NPAD, (long)NPAD * NPAD, NPAD,
                vt16, (long)2 * HD * NPAD, (long)HD * NPAD, NPAD,
                nullptr, 0, outf, (long)HN * HH, HD, HH,
                outf, nullptr, (long)HN * HH, HD, HH,
                HN, HD, NPAD, 2 * HB, 2, 1.f, 0);
    lnorm(outf, r_ln2_g + ib * HH, r_ln2_b + ib * HH, ln16, HBT);
    launch_gemm(stream, ln16, 0, 0, HH, rw116, 0, 0, HH,
                r_b1 + (size_t)ib * 4 * HH, 0, nullptr, 0, 0, 0,
                nullptr, mh16, 0, 0, 4 * HH, HBT, 4 * HH, HH, 1, 1, 1.f, 1);
    launch_gemm(stream, mh16, 0, 0, 4 * HH, rw216, 0, 0, 4 * HH,
                r_b2 + (size_t)ib * HH, 0, outf, 0, 0, HH,
                outf, nullptr, 0, 0, HH, HBT, HH, 4 * HH, 1, 1, 1.f, 0);
  }

  // ---- head ----
  k_head<<<1, 512, 0, stream>>>(outf, head_w, head_b, (float*)d_out);
}
